// SpectralAttention_59785944760343
// MI455X (gfx1250) — matrix operand order repaired and checked by reading, not compiled, not run on hardware
//
#include <hip/hip_runtime.h>
#include <stdint.h>

// ---------------- problem constants ----------------
constexpr int Bdim = 4;
constexpr int Ldim = 4096;
constexpr int Ddim = 2048;
constexpr int Mdim = Bdim * Ldim;   // 16384 rows
constexpr int Kdim = Ddim;          // 2048
constexpr int N1   = 3 * Ddim;      // 6144 (qkv)
constexpr int N2   = Ddim;          // 2048 (out proj)
constexpr int NCHUNK = 16;
constexpr int CL     = Ldim / NCHUNK; // 256

typedef __attribute__((ext_vector_type(16))) __bf16 v16bf;
typedef __attribute__((ext_vector_type(8)))  float  v8f;
typedef __attribute__((ext_vector_type(4)))  unsigned int u32x4;
typedef __attribute__((ext_vector_type(8)))  int          i32x8;
typedef __attribute__((ext_vector_type(4)))  int          i32x4;

#if defined(__has_builtin)
#if __has_builtin(__builtin_amdgcn_tensor_load_to_lds) && \
    __has_builtin(__builtin_amdgcn_s_wait_tensorcnt)
#define USE_TDM 1
#endif
#endif

// ---------------- helpers ----------------
__device__ __forceinline__ uint16_t f2bf(float f) {
  uint32_t u = __float_as_uint(f);
  u += 0x7fffu + ((u >> 16) & 1u);   // round to nearest even
  return (uint16_t)(u >> 16);
}
__device__ __forceinline__ float bf2f(uint32_t h) {
  return __uint_as_float(h << 16);
}
__device__ __forceinline__ float phi_f(float x) {
  // elu(x)+1 : x>0 -> x+1 ; x<=0 -> exp(x)
  return x > 0.f ? x + 1.f : __expf(x);
}

// ---------------- f32 -> bf16 conversion (vectorized x4) ----------------
__global__ __launch_bounds__(256) void cvt_f32_bf16_kernel(
    const float* __restrict__ in, uint16_t* __restrict__ out, int n4) {
  int i = blockIdx.x * 256 + threadIdx.x;
  if (i >= n4) return;
  float4 f = ((const float4*)in)[i];
  uint32_t lo = (uint32_t)f2bf(f.x) | ((uint32_t)f2bf(f.y) << 16);
  uint32_t hi = (uint32_t)f2bf(f.z) | ((uint32_t)f2bf(f.w) << 16);
  ((uint2*)out)[i] = make_uint2(lo, hi);
}

// ---------------- WMMA NT GEMM: C[M,N] = A[M,K] * B[N,K]^T ----------------
union FragBF { uint4 u[2]; v16bf v; };

constexpr int BM = 128, BN = 128, BK = 64;
constexpr int LDT = BK + 8;   // +16B pad per row: disjoint LDS banks for b128 frag reads

#if defined(USE_TDM)
// TDM descriptor: 2D tile, tile_dim=(64,128) elems of 2B, row stride = K elems,
// LDS padding: 16B (4 dwords, code 3) after every 128B (32 dwords, code 4) row.
__device__ __forceinline__ void tdm_load_tile(uint32_t lds_addr,
                                              const uint16_t* gptr, int kstride) {
  uint64_t ga = (uint64_t)(uintptr_t)gptr;
  u32x4 g0 = { 1u,                                    // count=1, valid user D#
               lds_addr,                              // LDS byte address
               (uint32_t)ga,                          // global_addr[31:0]
               (((uint32_t)(ga >> 32)) & 0x01FFFFFFu) // global_addr[56:32]
                 | 0x80000000u };                     // type=2 ("image")
  i32x8 g1 = { (int)((1u << 16)    // data_size = 2 bytes
                   | (1u << 20)    // pad_enable
                   | (4u << 22)    // pad_interval = 32 dwords (one 128B row)
                   | (3u << 25)),  // pad_amount   = 4 dwords (16B)
               (int)(64u << 16),                    // tensor_dim0 = 64
               (int)(128u << 16),                   // dim0_hi=0 | tensor_dim1 = 128
               (int)(64u << 16),                    // dim1_hi=0 | tile_dim0 = 64
               128,                                 // tile_dim1 = 128, tile_dim2 = 0
               kstride,                             // tensor_dim0_stride lo (elems)
               0, 0 };
  i32x4 gz = {0, 0, 0, 0};
#if __clang_major__ >= 23
  i32x8 g4 = {0, 0, 0, 0, 0, 0, 0, 0};
  __builtin_amdgcn_tensor_load_to_lds(g0, g1, gz, gz, g4, 0);
#else
  __builtin_amdgcn_tensor_load_to_lds(g0, g1, gz, gz, 0);
#endif
}
#endif

__device__ __forceinline__ void compute_tile(const uint16_t* __restrict__ sAb,
                                             const uint16_t* __restrict__ sBb,
                                             int wm, int wn, int fm, int kh,
                                             v8f (&acc)[2][4]) {
#pragma unroll
  for (int ks = 0; ks < 2; ++ks) {          // two 16x16x32 steps per BK=64
    FragBF fa[2], fb[4];
    // A 16x32 frag: lanes<16 -> K[0..7]+K[16..23]; lanes>=16 -> K[8..15]+K[24..31]
#pragma unroll
    for (int mt = 0; mt < 2; ++mt) {
      const uint16_t* p = &sAb[(wm * 32 + mt * 16 + fm) * LDT + ks * 32 + kh * 8];
      fa[mt].u[0] = *(const uint4*)p;
      fa[mt].u[1] = *(const uint4*)(p + 16);
    }
    // B 32x16 frag: lanes<16 -> K[0..7]+K[16..23]; lanes>=16 -> K[8..15]+K[24..31]
#pragma unroll
    for (int nt = 0; nt < 4; ++nt) {
      const uint16_t* p = &sBb[(wn * 64 + nt * 16 + fm) * LDT + ks * 32 + kh * 8];
      fb[nt].u[0] = *(const uint4*)p;
      fb[nt].u[1] = *(const uint4*)(p + 16);
    }
#pragma unroll
    for (int mt = 0; mt < 2; ++mt)
#pragma unroll
      for (int nt = 0; nt < 4; ++nt)
        acc[mt][nt] = __builtin_amdgcn_wmma_f32_16x16x32_bf16(
            false, fa[mt].v, false, fb[nt].v, (short)0, acc[mt][nt], false, false);
  }
}

template <bool OUT_BF16>
__global__ __launch_bounds__(256, 2) void gemm_nt_bf16_kernel(
    const uint16_t* __restrict__ A, const uint16_t* __restrict__ Bm,
    void* __restrict__ C, int Mv, int Nv, int Kv, float scale) {
  __shared__ __align__(16) uint16_t sA[2][BM * LDT];
  __shared__ __align__(16) uint16_t sB[2][BN * LDT];

  const int tid  = threadIdx.x;
  const int lane = tid & 31;
  const int wave = tid >> 5;
  const int wm   = wave >> 1;               // 0..3  (32-row slabs)
  const int wn   = wave & 1;                // 0..1  (64-col slabs)
  const int m0   = blockIdx.y * BM;
  const int n0   = blockIdx.x * BN;
  const int fm   = lane & 15;
  const int kh   = lane >> 4;
  const int nIter = Kv / BK;

  v8f acc[2][4];
#pragma unroll
  for (int i = 0; i < 2; ++i)
#pragma unroll
    for (int j = 0; j < 4; ++j)
#pragma unroll
      for (int e = 0; e < 8; ++e) acc[i][j][e] = 0.f;

#if defined(USE_TDM)
  // ---- Tensor Data Mover pipeline: wave 0 drives DMA, all waves compute ----
  const uint16_t* gA0 = A  + (size_t)m0 * Kv;
  const uint16_t* gB0 = Bm + (size_t)n0 * Kv;
  const uint32_t ldsA[2] = { (uint32_t)(uintptr_t)&sA[0][0],
                             (uint32_t)(uintptr_t)&sA[1][0] };
  const uint32_t ldsB[2] = { (uint32_t)(uintptr_t)&sB[0][0],
                             (uint32_t)(uintptr_t)&sB[1][0] };
  if (wave == 0) {                          // TDM ignores EXEC: branch-guard it
    tdm_load_tile(ldsA[0], gA0, Kv);
    tdm_load_tile(ldsB[0], gB0, Kv);
  }
  for (int it = 0; it < nIter; ++it) {
    if (wave == 0) {
      if (it + 1 < nIter) {                 // prefetch next tile into other buffer
        tdm_load_tile(ldsA[(it + 1) & 1], gA0 + (size_t)(it + 1) * BK, Kv);
        tdm_load_tile(ldsB[(it + 1) & 1], gB0 + (size_t)(it + 1) * BK, Kv);
        __builtin_amdgcn_s_wait_tensorcnt(2);   // current tile's 2 DMAs done
      } else {
        __builtin_amdgcn_s_wait_tensorcnt(0);
      }
    }
    __syncthreads();                        // publish DMA'd tile to all waves
    compute_tile(&sA[it & 1][0], &sB[it & 1][0], wm, wn, fm, kh, acc);
    __syncthreads();                        // all done reading before reuse
  }
#else
  // ---- fallback: manual double-buffered staging through VGPRs ----
  const int lrow = tid >> 1;                // 0..127
  const int lcol = (tid & 1) * 32;          // 0 or 32 elements
  const uint16_t* gA = A  + (size_t)(m0 + lrow) * Kv + lcol;
  const uint16_t* gB = Bm + (size_t)(n0 + lrow) * Kv + lcol;
  uint4 ra[4], rb[4];
#pragma unroll
  for (int j = 0; j < 4; ++j) {
    ra[j] = *(const uint4*)(gA + j * 8);
    rb[j] = *(const uint4*)(gB + j * 8);
  }
  {
    uint16_t* pa = &sA[0][lrow * LDT + lcol];
    uint16_t* pb = &sB[0][lrow * LDT + lcol];
#pragma unroll
    for (int j = 0; j < 4; ++j) {
      *(uint4*)(pa + j * 8) = ra[j];
      *(uint4*)(pb + j * 8) = rb[j];
    }
  }
  __syncthreads();
  for (int it = 0; it < nIter; ++it) {
    if (it + 1 < nIter) {
#pragma unroll
      for (int j = 0; j < 4; ++j) {         // issue next-tile loads before compute
        ra[j] = *(const uint4*)(gA + (size_t)(it + 1) * BK + j * 8);
        rb[j] = *(const uint4*)(gB + (size_t)(it + 1) * BK + j * 8);
      }
    }
    compute_tile(&sA[it & 1][0], &sB[it & 1][0], wm, wn, fm, kh, acc);
    if (it + 1 < nIter) {
      __syncthreads();
      uint16_t* pa = &sA[(it + 1) & 1][lrow * LDT + lcol];
      uint16_t* pb = &sB[(it + 1) & 1][lrow * LDT + lcol];
#pragma unroll
      for (int j = 0; j < 4; ++j) {
        *(uint4*)(pa + j * 8) = ra[j];
        *(uint4*)(pb + j * 8) = rb[j];
      }
      __syncthreads();
    }
  }
#endif

  // Epilogue per ISA C layout: VGPR r -> M = r + 8*(lane>=16), N = lane&15
  const int crow = m0 + wm * 32;
  const int ccol = n0 + wn * 64 + fm;
#pragma unroll
  for (int mt = 0; mt < 2; ++mt) {
#pragma unroll
    for (int nt = 0; nt < 4; ++nt) {
      const int row = crow + mt * 16 + kh * 8;
      const int col = ccol + nt * 16;
#pragma unroll
      for (int r = 0; r < 8; ++r) {
        float val = acc[mt][nt][r] * scale;
        size_t off = (size_t)(row + r) * Nv + col;
        if constexpr (OUT_BF16) ((uint16_t*)C)[off] = f2bf(val);
        else                    ((float*)C)[off]    = val;
      }
    }
  }
}

// ---------------- linear-attention scan (two-pass chunked prefix sum) ------
// qkv layout: [b*L + l][6144] with q at d, k at D+d, v at 2D+d.
// Each thread handles two adjacent channels -> dword-granular memory ops.
__global__ __launch_bounds__(256) void scan_partial_kernel(
    const uint16_t* __restrict__ qkv, float* __restrict__ pk,
    float* __restrict__ pkv) {
  int idx = blockIdx.x * 256 + threadIdx.x;   // [0, B*NCHUNK*D/2)
  int dp = idx & (Ddim / 2 - 1);
  int t  = idx >> 10;
  int c  = t & (NCHUNK - 1);
  int b  = t >> 4;
  int d  = dp * 2;
  const uint16_t* base = qkv + ((size_t)(b * Ldim + c * CL)) * 3 * Ddim + d;
  float sk0 = 0.f, sk1 = 0.f, skv0 = 0.f, skv1 = 0.f;
  for (int l = 0; l < CL; ++l) {
    uint32_t kk = *(const uint32_t*)(base + (size_t)l * 3 * Ddim + Ddim);
    uint32_t vv = *(const uint32_t*)(base + (size_t)l * 3 * Ddim + 2 * Ddim);
    float p0 = phi_f(bf2f(kk & 0xffffu));
    float p1 = phi_f(bf2f(kk >> 16));
    sk0 += p0; sk1 += p1;
    skv0 += p0 * bf2f(vv & 0xffffu);
    skv1 += p1 * bf2f(vv >> 16);
  }
  size_t o = (size_t)t * Ddim + d;
  pk[o] = sk0;  pk[o + 1] = sk1;
  pkv[o] = skv0; pkv[o + 1] = skv1;
}

__global__ __launch_bounds__(256) void scan_apply_kernel(
    const uint16_t* __restrict__ qkv, const float* __restrict__ pk,
    const float* __restrict__ pkv, uint16_t* __restrict__ attn) {
  int idx = blockIdx.x * 256 + threadIdx.x;
  int dp = idx & (Ddim / 2 - 1);
  int t  = idx >> 10;
  int c  = t & (NCHUNK - 1);
  int b  = t >> 4;
  int d  = dp * 2;
  float sk0 = 0.f, sk1 = 0.f, skv0 = 0.f, skv1 = 0.f;
  for (int cc = 0; cc < c; ++cc) {            // exclusive prefix of chunk totals
    size_t o = (size_t)(b * NCHUNK + cc) * Ddim + d;
    sk0 += pk[o];  sk1 += pk[o + 1];
    skv0 += pkv[o]; skv1 += pkv[o + 1];
  }
  const uint16_t* base = qkv + ((size_t)(b * Ldim + c * CL)) * 3 * Ddim + d;
  uint16_t* ob = attn + (size_t)(b * Ldim + c * CL) * Ddim + d;
  for (int l = 0; l < CL; ++l) {
    uint32_t qq = *(const uint32_t*)(base + (size_t)l * 3 * Ddim);
    uint32_t kk = *(const uint32_t*)(base + (size_t)l * 3 * Ddim + Ddim);
    uint32_t vv = *(const uint32_t*)(base + (size_t)l * 3 * Ddim + 2 * Ddim);
    float p0 = phi_f(bf2f(kk & 0xffffu));
    float p1 = phi_f(bf2f(kk >> 16));
    skv0 += p0 * bf2f(vv & 0xffffu);
    skv1 += p1 * bf2f(vv >> 16);
    sk0 += p0; sk1 += p1;
    float pq0 = phi_f(bf2f(qq & 0xffffu));
    float pq1 = phi_f(bf2f(qq >> 16));
    float o0 = pq0 * skv0 / (pq0 * sk0 + 1e-9f);
    float o1 = pq1 * skv1 / (pq1 * sk1 + 1e-9f);
    *(uint32_t*)(ob + (size_t)l * Ddim) =
        (uint32_t)f2bf(o0) | ((uint32_t)f2bf(o1) << 16);
  }
}

// ---------------- launcher ----------------
extern "C" void kernel_launch(void* const* d_in, const int* in_sizes, int n_in,
                              void* d_out, int out_size, void* d_ws, size_t ws_size,
                              hipStream_t stream) {
  (void)in_sizes; (void)n_in; (void)out_size; (void)ws_size;
  const float* x    = (const float*)d_in[0];   // (B,L,D)
  const float* Wqkv = (const float*)d_in[1];   // (3D,D)
  const float* Wout = (const float*)d_in[2];   // (D,D)

  const size_t nX    = (size_t)Mdim * Kdim;    // 33.5M
  const size_t nWq   = (size_t)N1 * Kdim;      // 12.6M
  const size_t nWo   = (size_t)N2 * Kdim;      //  4.2M
  const size_t nQKV  = (size_t)Mdim * N1;      // 100.7M
  const size_t nAttn = (size_t)Mdim * Ddim;    // 33.5M
  const size_t nPart = (size_t)Bdim * NCHUNK * Ddim;  // 131072

  uint16_t* xb    = (uint16_t*)d_ws;
  uint16_t* wqb   = xb + nX;
  uint16_t* wob   = wqb + nWq;
  uint16_t* qkvb  = wob + nWo;
  uint16_t* attnb = qkvb + nQKV;
  float*    pk    = (float*)(attnb + nAttn);
  float*    pkv   = pk + nPart;

  // 1) cast inputs to bf16
  cvt_f32_bf16_kernel<<<(int)(nX / 4 / 256), 256, 0, stream>>>(x, xb, (int)(nX / 4));
  cvt_f32_bf16_kernel<<<(int)(nWq / 4 / 256), 256, 0, stream>>>(Wqkv, wqb, (int)(nWq / 4));
  cvt_f32_bf16_kernel<<<(int)(nWo / 4 / 256), 256, 0, stream>>>(Wout, wob, (int)(nWo / 4));

  // 2) qkv = x @ Wqkv^T  (bf16 WMMA, f32 accum, bf16 out)
  dim3 g1(N1 / 128, Mdim / 128);
  gemm_nt_bf16_kernel<true><<<g1, 256, 0, stream>>>(xb, wqb, (void*)qkvb,
                                                    Mdim, N1, Kdim, 1.0f);

  // 3) linear-attention chunked scan (2 channels per thread)
  const int nscan2 = (int)(nPart / 2);
  scan_partial_kernel<<<nscan2 / 256, 256, 0, stream>>>(qkvb, pk, pkv);
  scan_apply_kernel<<<nscan2 / 256, 256, 0, stream>>>(qkvb, pk, pkv, attnb);

  // 4) out = attn @ Wout^T * HD^-0.5  (f32 output)
  dim3 g2(N2 / 128, Mdim / 128);
  gemm_nt_bf16_kernel<false><<<g2, 256, 0, stream>>>(attnb, wob, d_out,
                                                     Mdim, N2, Kdim,
                                                     0.08838834764831845f);
}
